// SleepSNN_15831249453649
// MI455X (gfx1250) — matrix operand order repaired and checked by reading, not compiled, not run on hardware
//
#include <hip/hip_runtime.h>
#include <hip/hip_bf16.h>
#include <stddef.h>

// CDNA5 / gfx1250: wave32, WMMA 16x16x32 f16 -> f32 accum.
typedef __attribute__((ext_vector_type(16))) _Float16 v16h;
typedef __attribute__((ext_vector_type(8)))  float    v8f;

#define B_SZ   256
#define T_SZ   100
#define NI_SZ  784
#define NH_SZ  1024         // 32 chunks of K=32
#define NO_SZ  10
#define NO_PAD 16
#define KC_IN  25           // ceil(784/32), zero-padded to 800
#define KC_REC 32
#define NT_H   64           // NH/16 hidden tiles
#define MT_B   16           // B/16 batch tiles

static __device__ __forceinline__ float clip2(float v) {
    return fminf(2.0f, fmaxf(-2.0f, v));
}

static __device__ __forceinline__ v8f wmma_f16(v16h a, v16h b, v8f c) {
    return __builtin_amdgcn_wmma_f32_16x16x32_f16(false, a, false, b,
                                                  (short)0, c, false, false);
}

// ---------------------------------------------------------------------------
// Weight packing: ISA 32x16 16-bit B layout. Lane L holds column n=(L&15),
// K = (e>>3)*16 + (L>>4)*8 + (e&7) (e=0..15)  -> one contiguous v16h (32B) per lane.
// ---------------------------------------------------------------------------
__global__ __launch_bounds__(256) void pack_w1(
    const float* __restrict__ wf, const float* __restrict__ ws,
    _Float16* __restrict__ out) {
    int idx = blockIdx.x * 256 + threadIdx.x;            // < KC_IN*NT_H*32*16
    if (idx >= KC_IN * NT_H * 32 * 16) return;
    int e    = idx & 15;
    int lane = (idx >> 4) & 31;
    int nt   = (idx >> 9) & 63;
    int kc   = idx >> 15;
    int n = nt * 16 + (lane & 15);
    int k = kc * 32 + ((e >> 3) * 16) + ((lane >> 4) * 8) + (e & 7);
    float v = 0.0f;
    if (k < NI_SZ) v = clip2(wf[(size_t)n * NI_SZ + k] + ws[(size_t)n * NI_SZ + k]);
    out[idx] = (_Float16)v;
}

__global__ __launch_bounds__(256) void pack_wrec(
    const float* __restrict__ wf, const float* __restrict__ ws,
    _Float16* __restrict__ out) {
    int idx = blockIdx.x * 256 + threadIdx.x;            // < KC_REC*NT_H*32*16
    if (idx >= KC_REC * NT_H * 32 * 16) return;
    int e    = idx & 15;
    int lane = (idx >> 4) & 31;
    int nt   = (idx >> 9) & 63;
    int kc   = idx >> 15;
    int n = nt * 16 + (lane & 15);
    int k = kc * 32 + ((e >> 3) * 16) + ((lane >> 4) * 8) + (e & 7);
    out[idx] = (_Float16)clip2(wf[(size_t)n * NH_SZ + k] + ws[(size_t)n * NH_SZ + k]);
}

__global__ __launch_bounds__(256) void pack_wout(
    const float* __restrict__ wf, const float* __restrict__ ws,
    _Float16* __restrict__ out) {
    int idx = blockIdx.x * 256 + threadIdx.x;            // < KC_REC*32*16
    if (idx >= KC_REC * 32 * 16) return;
    int e    = idx & 15;
    int lane = (idx >> 4) & 31;
    int kc   = idx >> 9;
    int n = lane & 15;                                   // N padded 10 -> 16
    int k = kc * 32 + ((e >> 3) * 16) + ((lane >> 4) * 8) + (e & 7);
    float v = 0.0f;
    if (n < NO_SZ) v = clip2(wf[(size_t)n * NH_SZ + k] + ws[(size_t)n * NH_SZ + k]);
    out[idx] = (_Float16)v;
}

// ---------------------------------------------------------------------------
// Per-step x_t -> f16 A-fragment pack:  [mt][kc][lane][e]
// A layout (16-bit 16x32): lane L holds row m=(L&15),
//   K(e) = (e>>3)*16 + (L>>4)*8 + (e&7).
// Converts each x element exactly once per step (spikes are exact in f16).
// ---------------------------------------------------------------------------
__global__ __launch_bounds__(256) void pack_x(
    const float* __restrict__ x, _Float16* __restrict__ xp, int t) {
    int idx = blockIdx.x * 256 + threadIdx.x;            // < MT_B*KC_IN*32*16
    if (idx >= MT_B * KC_IN * 32 * 16) return;
    int e    = idx & 15;
    int lane = (idx >> 4) & 31;
    int rest = idx >> 9;
    int kc   = rest % KC_IN;
    int mt   = rest / KC_IN;
    int m = mt * 16 + (lane & 15);
    int k = kc * 32 + ((e >> 3) * 16) + ((lane >> 4) * 8) + (e & 7);
    float v = 0.0f;
    if (k < NI_SZ) v = x[((size_t)m * T_SZ + t) * NI_SZ + k];
    xp[idx] = (_Float16)v;
}

// Zero all persistent state (harness poisons ws with 0xAA each run).
__global__ __launch_bounds__(256) void init_state(
    float* __restrict__ v_h, float* __restrict__ a_h,
    _Float16* __restrict__ spk0, _Float16* __restrict__ spk1,
    float* __restrict__ v_out, float* __restrict__ out_sum) {
    int i = blockIdx.x * 256 + threadIdx.x;              // < B*NH
    v_h[i] = 0.0f; a_h[i] = 0.0f;
    spk0[i] = (_Float16)0.0f; spk1[i] = (_Float16)0.0f;
    if (i < B_SZ * NO_PAD) { v_out[i] = 0.0f; out_sum[i] = 0.0f; }
}

// ---------------------------------------------------------------------------
// Hidden layer step: each wave computes a 2x2 block of 16x16 tiles (32x32
// outputs): per K-chunk load 2 A + 2 B fragments, issue 4 WMMAs (4
// independent accumulator chains hide WMMA->WMMA RAW latency, 2x fewer
// loads per WMMA). Grid: 8 mt-pairs x 32 nt-pairs = 256 waves.
//   I_h = x_t @ w1^T + spk_prev @ w_rec^T, then fused adaptive-LIF update.
// C/D layout: elem r at lane L -> (m = r + 8*(L>>4), n = L&15).
// ---------------------------------------------------------------------------
__global__ __launch_bounds__(256) void snn_hidden_step(
    const _Float16* __restrict__ xp,      // [MT_B][KC_IN][32][16]
    const _Float16* __restrict__ w1p,     // [KC_IN][NT_H][32][16]
    const _Float16* __restrict__ wrp,     // [KC_REC][NT_H][32][16]
    const float* __restrict__ alpha,      // (NH)
    const _Float16* __restrict__ spk_prev,// (B, NH)
    _Float16* __restrict__ spk_next,      // (B, NH)
    float* __restrict__ v_h,              // (B, NH)
    float* __restrict__ a_h) {            // (B, NH)
    const int lane  = threadIdx.x & 31;
    const int wave  = threadIdx.x >> 5;
    const int wid   = blockIdx.x * 8 + wave;   // 0..255
    const int mt2   = wid >> 5;                // 0..7  (pair of batch tiles)
    const int nt2   = wid & 31;                // 0..31 (pair of hidden tiles)
    const int mt0   = mt2 * 2, mt1 = mt0 + 1;
    const int nt0   = nt2 * 2, nt1 = nt0 + 1;
    const int lhalf = lane >> 4;
    const int lmod  = lane & 15;
    const int ka    = lhalf * 8;

    v8f acc00 = {0.f,0.f,0.f,0.f,0.f,0.f,0.f,0.f};
    v8f acc01 = acc00, acc10 = acc00, acc11 = acc00;

    // ---- input contribution: packed f16 A fragments, contiguous 32B/lane ----
    const _Float16* xa0 = xp + ((size_t)mt0 * KC_IN * 32 + lane) * 16;
    const _Float16* xa1 = xp + ((size_t)mt1 * KC_IN * 32 + lane) * 16;
    for (int kc = 0; kc < KC_IN; ++kc) {
        v16h a0 = *(const v16h*)(xa0 + (size_t)kc * 32 * 16);
        v16h a1 = *(const v16h*)(xa1 + (size_t)kc * 32 * 16);
        const _Float16* bp = w1p + (((size_t)kc * NT_H + nt0) * 32 + lane) * 16;
        __builtin_prefetch(bp + (size_t)NT_H * 32 * 16, 0, 1);  // next K-chunk
        v16h b0 = *(const v16h*)bp;
        v16h b1 = *(const v16h*)(bp + 32 * 16);                 // nt1 fragment
        acc00 = wmma_f16(a0, b0, acc00);
        acc01 = wmma_f16(a0, b1, acc01);
        acc10 = wmma_f16(a1, b0, acc10);
        acc11 = wmma_f16(a1, b1, acc11);
    }

    // ---- recurrent contribution: spk_prev (f16, exact {0,1}) ----
    const _Float16* s0 = spk_prev + (size_t)(mt0 * 16 + lmod) * NH_SZ;
    const _Float16* s1 = spk_prev + (size_t)(mt1 * 16 + lmod) * NH_SZ;
    for (int kc = 0; kc < KC_REC; ++kc) {
        v16h a0, a1;
#pragma unroll
        for (int e = 0; e < 16; ++e) {
            int k = kc * 32 + ((e >> 3) * 16) + ka + (e & 7);   // 2x 16B runs
            a0[e] = s0[k];
            a1[e] = s1[k];
        }
        const _Float16* bp = wrp + (((size_t)kc * NT_H + nt0) * 32 + lane) * 16;
        __builtin_prefetch(bp + (size_t)NT_H * 32 * 16, 0, 1);
        v16h b0 = *(const v16h*)bp;
        v16h b1 = *(const v16h*)(bp + 32 * 16);
        acc00 = wmma_f16(a0, b0, acc00);
        acc01 = wmma_f16(a0, b1, acc01);
        acc10 = wmma_f16(a1, b0, acc10);
        acc11 = wmma_f16(a1, b1, acc11);
    }

    // ---- fused adaptive-LIF update over the 2x2 tile block ----
#pragma unroll
    for (int ni = 0; ni < 2; ++ni) {
        const int h = (nt0 + ni) * 16 + lmod;
        const float al = alpha[h];
#pragma unroll
        for (int mi = 0; mi < 2; ++mi) {
            const v8f acc = (ni == 0) ? (mi == 0 ? acc00 : acc10)
                                      : (mi == 0 ? acc01 : acc11);
            const int m0 = (mt0 + mi) * 16;
#pragma unroll
            for (int r = 0; r < 8; ++r) {
                int bb = m0 + r + lhalf * 8;
                size_t idx = (size_t)bb * NH_SZ + h;
                float sp = (float)spk_prev[idx];
                float vh = al * v_h[idx] + acc[r] - sp;       // subtractive reset
                float ah = 0.9f * a_h[idx] + sp;              // RHO adaptation
                float s  = (vh - (1.0f + 0.1f * ah)) > 0.0f ? 1.0f : 0.0f;
                v_h[idx] = vh;
                a_h[idx] = ah;
                spk_next[idx] = (_Float16)s;
            }
        }
    }
}

// ---------------------------------------------------------------------------
// Output layer step: 16 waves, one per batch tile. N padded to 16.
//   I_out = spk_new @ w_out^T ; output-LIF ; running sum ; final /T write.
// ---------------------------------------------------------------------------
__global__ __launch_bounds__(512) void snn_out_step(
    const _Float16* __restrict__ spk,     // new spikes (B, NH)
    const _Float16* __restrict__ wop,     // [KC_REC][32][16]
    const float* __restrict__ beta,       // (NO)
    float* __restrict__ v_out,            // (B, NO_PAD)
    float* __restrict__ out_sum,          // (B, NO_PAD)
    float* __restrict__ out,              // (B, NO) final
    int is_last) {
    const int lane  = threadIdx.x & 31;
    const int wave  = threadIdx.x >> 5;   // batch tile 0..15
    const int m0    = wave * 16;
    const int lhalf = lane >> 4;
    const int lmod  = lane & 15;
    const int ka    = lhalf * 8;

    v8f acc = {0.f,0.f,0.f,0.f,0.f,0.f,0.f,0.f};
    const _Float16* srow = spk + (size_t)(m0 + lmod) * NH_SZ;
    for (int kc = 0; kc < KC_REC; ++kc) {
        v16h a;
#pragma unroll
        for (int e = 0; e < 16; ++e) {
            int k = kc * 32 + ((e >> 3) * 16) + ka + (e & 7);
            a[e] = srow[k];
        }
        v16h b = *(const v16h*)(wop + ((size_t)kc * 32 + lane) * 16);
        acc = wmma_f16(a, b, acc);
    }

    const int o = lmod;
    const float bt = (o < NO_SZ) ? beta[o] : 0.0f;
#pragma unroll
    for (int r = 0; r < 8; ++r) {
        int bb = m0 + r + lhalf * 8;
        int idx = bb * NO_PAD + o;
        float vo = bt * v_out[idx] + acc[r];
        float so = (vo - 1.0f) > 0.0f ? 1.0f : 0.0f;
        vo -= so;                                          // subtractive reset
        float os = out_sum[idx] + vo;
        v_out[idx]  = vo;
        out_sum[idx] = os;
        if (is_last && o < NO_SZ) out[bb * NO_SZ + o] = os * (1.0f / (float)T_SZ);
    }
}

// ---------------------------------------------------------------------------
extern "C" void kernel_launch(void* const* d_in, const int* in_sizes, int n_in,
                              void* d_out, int out_size, void* d_ws, size_t ws_size,
                              hipStream_t stream) {
    const float* x          = (const float*)d_in[0];
    const float* w1_fast    = (const float*)d_in[1];
    const float* w_rec_fast = (const float*)d_in[2];
    const float* w_out_fast = (const float*)d_in[3];
    const float* fc1_slow   = (const float*)d_in[4];
    const float* fc_rec_slow= (const float*)d_in[5];
    const float* fc_out_slow= (const float*)d_in[6];
    const float* alpha      = (const float*)d_in[7];
    const float* beta       = (const float*)d_in[8];
    float* out = (float*)d_out;

    // Workspace layout (all chunks 32B-aligned; total ~7.1 MB).
    char* p = (char*)d_ws;
    _Float16* w1p = (_Float16*)p; p += (size_t)KC_IN  * NT_H * 32 * 16 * 2;
    _Float16* wrp = (_Float16*)p; p += (size_t)KC_REC * NT_H * 32 * 16 * 2;
    _Float16* wop = (_Float16*)p; p += (size_t)KC_REC * 32 * 16 * 2;
    _Float16* xpk = (_Float16*)p; p += (size_t)MT_B * KC_IN * 32 * 16 * 2;
    float* v_h    = (float*)p;    p += (size_t)B_SZ * NH_SZ * 4;
    float* a_h    = (float*)p;    p += (size_t)B_SZ * NH_SZ * 4;
    _Float16* spk0= (_Float16*)p; p += (size_t)B_SZ * NH_SZ * 2;
    _Float16* spk1= (_Float16*)p; p += (size_t)B_SZ * NH_SZ * 2;
    float* v_out  = (float*)p;    p += (size_t)B_SZ * NO_PAD * 4;
    float* o_sum  = (float*)p;    p += (size_t)B_SZ * NO_PAD * 4;
    _Float16* spk[2] = { spk0, spk1 };

    // One-time (per launch) weight packing + state init.
    pack_w1  <<<(KC_IN  * NT_H * 32 * 16 + 255) / 256, 256, 0, stream>>>(w1_fast,    fc1_slow,    w1p);
    pack_wrec<<<(KC_REC * NT_H * 32 * 16 + 255) / 256, 256, 0, stream>>>(w_rec_fast, fc_rec_slow, wrp);
    pack_wout<<<(KC_REC * 32 * 16        + 255) / 256, 256, 0, stream>>>(w_out_fast, fc_out_slow, wop);
    init_state<<<(B_SZ * NH_SZ + 255) / 256, 256, 0, stream>>>(v_h, a_h, spk0, spk1, v_out, o_sum);

    // Sequential time loop: spikes double-buffered across steps.
    for (int t = 0; t < T_SZ; ++t) {
        _Float16* sp_prev = spk[t & 1];
        _Float16* sp_next = spk[(t + 1) & 1];
        pack_x<<<(MT_B * KC_IN * 32 * 16 + 255) / 256, 256, 0, stream>>>(x, xpk, t);
        snn_hidden_step<<<32, 256, 0, stream>>>(
            xpk, w1p, wrp, alpha, sp_prev, sp_next, v_h, a_h);
        snn_out_step<<<1, 512, 0, stream>>>(
            sp_next, wop, beta, v_out, o_sum, out, (t == T_SZ - 1) ? 1 : 0);
    }
}
